// BUNet_59545426592079
// MI455X (gfx1250) — matrix operand order repaired and checked by reading, compiled, not run on hardware
//
#include <hip/hip_runtime.h>
#include <cstdint>
#include <cstddef>

// ---------------------------------------------------------------------------
// Types for CDNA5 WMMA (wave32): v_wmma_f32_16x16x32_bf16
// ---------------------------------------------------------------------------
typedef __bf16 bf16_t;
typedef __attribute__((ext_vector_type(8)))  __bf16 bf16x8;
typedef __attribute__((ext_vector_type(16))) __bf16 bf16x16;
typedef __attribute__((ext_vector_type(8)))  float  f32x8;

#define N_MOL 65536
#define E_MOL 131072
#define B_MOL 2048
#define N_PRO 300000
#define E_PRO 1200000
#define B_PRO 2000
#define E_PPI 64000

__device__ __forceinline__ bf16_t f2bf(float f) {
  union { float f; uint32_t u; } v; v.f = f;
  uint32_t r = v.u + 0x7FFFu + ((v.u >> 16) & 1u);   // round-to-nearest-even
  union { uint16_t u; bf16_t b; } o; o.u = (uint16_t)(r >> 16);
  return o.b;
}

// ---------------------------------------------------------------------------
// Elementwise / setup kernels
// ---------------------------------------------------------------------------
__global__ void k_fill(float* __restrict__ p, float v, long long n) {
  long long i = (long long)blockIdx.x * blockDim.x + threadIdx.x;
  if (i < n) p[i] = v;
}

__global__ void k_deg(const int* __restrict__ dst, float* __restrict__ deg, int E) {
  int e = blockIdx.x * blockDim.x + threadIdx.x;
  if (e < E) atomicAdd(&deg[dst[e]], 1.0f);
}

__global__ void k_rsqrt(float* __restrict__ p, int n) {
  int i = blockIdx.x * blockDim.x + threadIdx.x;
  if (i < n) p[i] = rsqrtf(p[i]);          // deg >= 1 always (self loop)
}

// f32 [M x K] -> bf16 [M x Kpad], zero-padded cols, optional input ReLU
__global__ void k_pack_act(const float* __restrict__ X, bf16_t* __restrict__ Xb,
                           long long M, int K, int Kpad, int relu) {
  long long idx = (long long)blockIdx.x * blockDim.x + threadIdx.x;
  if (idx >= M * (long long)Kpad) return;
  long long i = idx / Kpad; int j = (int)(idx % Kpad);
  float v = (j < K) ? X[i * (long long)K + j] : 0.0f;
  if (relu) v = fmaxf(v, 0.0f);
  Xb[idx] = f2bf(v);
}

// Pack f32 weight [K x N] into WMMA B-fragment order:
// blob[(nt*Ktiles + kt)*512 + lane*16 + e], lane holds column n = nt*16+(lane&15),
// K index = kt*32 + (lane<16 ? 0 : 16) + e   (contiguous k order, matching the A loads).
// Ntiles is padded to a multiple of 8 so the GEMM needs no N-tile load guards.
__global__ void k_pack_wfrag(const float* __restrict__ W, bf16_t* __restrict__ Wb,
                             int K, int N, int Ktiles, int Ntiles) {
  long long idx = (long long)blockIdx.x * blockDim.x + threadIdx.x;
  long long total = (long long)Ktiles * Ntiles * 512;
  if (idx >= total) return;
  int e = (int)(idx & 15);
  int l = (int)((idx >> 4) & 31);
  long long t = idx >> 9;
  int kt = (int)(t % Ktiles);
  int nt = (int)(t / Ktiles);
  int n = nt * 16 + (l & 15);
  int k = kt * 32 + ((l < 16) ? 0 : 16) + e;
  float v = (k < K && n < N) ? W[(size_t)k * N + n] : 0.0f;
  Wb[idx] = f2bf(v);
}

// ---------------------------------------------------------------------------
// WMMA GEMM:  C[MxN](f32) = A[MxKpad](bf16, row-major) * Bfrag  (+bias, +relu)
// One wave computes a 16x128 strip: 8 N-tiles / 8 f32x8 accumulators, so each
// A-fragment load pair feeds 8 back-to-back WMMAs (disjoint D regs -> no RAW
// hazard stalls). 4 waves/block cover 4 consecutive M-tiles. Requires M%16==0
// (65536 / 300000 / 2048 / 2000 all are). Bfrag is padded to 8-tile multiples.
// ---------------------------------------------------------------------------
__global__ void __launch_bounds__(128) k_wmma_gemm(
    const bf16_t* __restrict__ A, int M, int Kpad,
    const bf16_t* __restrict__ Bfrag, int Ktiles,
    const float* __restrict__ bias,
    float* __restrict__ C, int N, int relu) {
  int lane = threadIdx.x & 31;
  int wave = threadIdx.x >> 5;
  int Mtiles = M >> 4;
  int tm = blockIdx.y * 4 + wave;
  if (tm >= Mtiles) return;                 // wave-uniform exit: EXEC all-1s for WMMA
  int tn0 = blockIdx.x * 8;
  int r = (tm << 4) + (lane & 15);
  int kbaseA = (lane < 16) ? 0 : 16;
  const bf16_t* arow  = A + (size_t)r * Kpad;
  const bf16_t* bbase = Bfrag + ((size_t)tn0 * Ktiles << 9) + (lane << 4);
  size_t bstep = (size_t)Ktiles << 9;       // elements between consecutive N-tiles

  f32x8 zero = {};
  f32x8 acc[8];
#pragma unroll
  for (int t = 0; t < 8; ++t) acc[t] = zero;

  for (int kt = 0; kt < Ktiles; ++kt) {
    bf16x8 a0 = *(const bf16x8*)(arow + kt * 32 + kbaseA);        // 16B load
    bf16x8 a1 = *(const bf16x8*)(arow + kt * 32 + kbaseA + 8);    // 16B load
    bf16x16 a = __builtin_shufflevector(a0, a1,
        0,1,2,3,4,5,6,7,8,9,10,11,12,13,14,15);
    const bf16_t* bk = bbase + ((size_t)kt << 9);
#pragma unroll
    for (int t = 0; t < 8; ++t) {
      bf16x8 b0 = *(const bf16x8*)(bk);
      bf16x8 b1 = *(const bf16x8*)(bk + 8);
      bf16x16 b = __builtin_shufflevector(b0, b1,
          0,1,2,3,4,5,6,7,8,9,10,11,12,13,14,15);
      acc[t] = __builtin_amdgcn_wmma_f32_16x16x32_bf16(
          false, a, false, b, (short)0, acc[t], false, false);
      bk += bstep;
    }
  }

  int rbase = (tm << 4) + ((lane < 16) ? 0 : 8);
#pragma unroll
  for (int t = 0; t < 8; ++t) {
    int c = ((tn0 + t) << 4) + (lane & 15);
    if (c < N) {
      float bv = bias ? bias[c] : 0.0f;
#pragma unroll
      for (int e = 0; e < 8; ++e) {
        float v = acc[t][e] + bv;
        if (relu) v = fmaxf(v, 0.0f);
        C[(size_t)(rbase + e) * N + c] = v;
      }
    }
  }
}

// ---------------------------------------------------------------------------
// GCN pieces: Out = b + H*dis^2 (self loop), then edge scatter atomics
// ---------------------------------------------------------------------------
__global__ void k_self_bias(const float* __restrict__ H, const float* __restrict__ dis,
                            const float* __restrict__ b, float* __restrict__ Out,
                            long long M, int N) {
  long long idx = (long long)blockIdx.x * blockDim.x + threadIdx.x;
  if (idx >= M * (long long)N) return;
  long long i = idx / N; int j = (int)(idx % N);
  float di = dis[i];
  Out[idx] = b[j] + H[idx] * di * di;
}

// float4-wide edge scatter (all N here are multiples of 4; rows 16B aligned)
__global__ void k_scatter4(const int* __restrict__ src, const int* __restrict__ dst,
                           const float* __restrict__ H, const float* __restrict__ dis,
                           float* __restrict__ Out, long long E, int N) {
  int N4 = N >> 2;
  long long idx = (long long)blockIdx.x * blockDim.x + threadIdx.x;
  if (idx >= E * (long long)N4) return;
  long long e = idx / N4; int j = (int)(idx % N4) << 2;
  int s = src[e], d = dst[e];
  float cf = dis[s] * dis[d];
  const float4 h = *(const float4*)(H + (size_t)s * N + j);
  float* o = Out + (size_t)d * N + j;
  atomicAdd(o + 0, h.x * cf);
  atomicAdd(o + 1, h.y * cf);
  atomicAdd(o + 2, h.z * cf);
  atomicAdd(o + 3, h.w * cf);
}

// segment-mean pooling with fused ReLU on the input read
__global__ void k_pool_accum(const float* __restrict__ X, const int* __restrict__ batch,
                             float* __restrict__ sums, float* __restrict__ cnt,
                             long long M, int N) {
  long long idx = (long long)blockIdx.x * blockDim.x + threadIdx.x;
  if (idx >= M * (long long)N) return;
  long long i = idx / N; int j = (int)(idx % N);
  int b = batch[i];
  atomicAdd(&sums[(size_t)b * N + j], fmaxf(X[idx], 0.0f));
  if (j == 0) atomicAdd(&cnt[b], 1.0f);
}

__global__ void k_pool_div(float* __restrict__ sums, const float* __restrict__ cnt,
                           long long B, int N) {
  long long idx = (long long)blockIdx.x * blockDim.x + threadIdx.x;
  if (idx >= B * (long long)N) return;
  long long b = idx / N;
  sums[idx] /= fmaxf(cnt[b], 1.0f);
}

// build bf16 [B x 256] = concat(x[i,0:128], q[seq[i],0:128])
__global__ void k_build_xc(const float* __restrict__ x, const float* __restrict__ q,
                           const int* __restrict__ seq, bf16_t* __restrict__ xc, int B) {
  int idx = blockIdx.x * blockDim.x + threadIdx.x;
  if (idx >= B * 256) return;
  int i = idx >> 8; int j = idx & 255;
  float v = (j < 128) ? x[(size_t)i * 128 + j]
                      : q[(size_t)seq[i] * 128 + (j - 128)];
  xc[idx] = f2bf(v);
}

// out[i] = h2[i,:] . w + b   (N=1 head; WMMA would waste 15/16 of the tile)
__global__ void k_head_out(const float* __restrict__ h, const float* __restrict__ w,
                           const float* __restrict__ b, float* __restrict__ out,
                           int B, int K) {
  int i = blockIdx.x * blockDim.x + threadIdx.x;
  if (i >= B) return;
  float s = b[0];
  const float* row = h + (size_t)i * K;
  for (int k = 0; k < K; ++k) s = fmaf(row[k], w[k], s);
  out[i] = s;
}

// ---------------------------------------------------------------------------
// Host driver
// ---------------------------------------------------------------------------
extern "C" void kernel_launch(void* const* d_in, const int* in_sizes, int n_in,
                              void* d_out, int out_size, void* d_ws, size_t ws_size,
                              hipStream_t stream) {
  (void)in_sizes; (void)n_in; (void)out_size; (void)ws_size;

  const float* mol_x   = (const float*)d_in[0];
  const float* pro_x   = (const float*)d_in[1];
  const float* w_mg1   = (const float*)d_in[2];
  const float* b_mg1   = (const float*)d_in[3];
  const float* w_mg2   = (const float*)d_in[4];
  const float* b_mg2   = (const float*)d_in[5];
  const float* w_mg3   = (const float*)d_in[6];
  const float* b_mg3   = (const float*)d_in[7];
  const float* w_mfc1  = (const float*)d_in[8];
  const float* b_mfc1  = (const float*)d_in[9];
  const float* w_mfc2  = (const float*)d_in[10];
  const float* b_mfc2  = (const float*)d_in[11];
  const float* w_pg1   = (const float*)d_in[12];
  const float* b_pg1   = (const float*)d_in[13];
  const float* w_pg2   = (const float*)d_in[14];
  const float* b_pg2   = (const float*)d_in[15];
  const float* w_pg3   = (const float*)d_in[16];
  const float* b_pg3   = (const float*)d_in[17];
  const float* w_pfc1  = (const float*)d_in[18];
  const float* b_pfc1  = (const float*)d_in[19];
  const float* w_pfc2  = (const float*)d_in[20];
  const float* b_pfc2  = (const float*)d_in[21];
  const float* w_ppig1 = (const float*)d_in[22];
  const float* b_ppig1 = (const float*)d_in[23];
  const float* w_ppig2 = (const float*)d_in[24];
  const float* b_ppig2 = (const float*)d_in[25];
  const float* w_ppifc1= (const float*)d_in[26];
  const float* b_ppifc1= (const float*)d_in[27];
  const float* w_ppifc2= (const float*)d_in[28];
  const float* b_ppifc2= (const float*)d_in[29];
  const float* w_fc1   = (const float*)d_in[30];
  const float* b_fc1   = (const float*)d_in[31];
  const float* w_fc2   = (const float*)d_in[32];
  const float* b_fc2   = (const float*)d_in[33];
  const float* w_outp  = (const float*)d_in[34];
  const float* b_outp  = (const float*)d_in[35];
  const int* mol_ei    = (const int*)d_in[36];
  const int* mol_batch = (const int*)d_in[37];
  const int* pro_ei    = (const int*)d_in[38];
  const int* pro_batch = (const int*)d_in[39];
  const int* ppi_e     = (const int*)d_in[40];
  const int* seq_num   = (const int*)d_in[41];

  // --- bump allocator over d_ws ---
  char* wsb = (char*)d_ws;
  size_t off = 0;
  auto alloc = [&](size_t bytes) -> void* {
    off = (off + 255) & ~(size_t)255;
    void* p = wsb + off;
    off += bytes;
    return p;
  };

  // big reusable buffers
  size_t actElems = (size_t)N_PRO * 128;                 // 38.4M
  size_t molActE  = (size_t)N_MOL * 320;                 // 21.0M
  if (molActE > actElems) actElems = molActE;
  bf16_t* actBig = (bf16_t*)alloc(actElems * sizeof(bf16_t));
  size_t hElems = (size_t)N_PRO * 128;                   // 38.4M f32 (covers mol 65536x320)
  float* Hbig = (float*)alloc(hElems * sizeof(float));
  float* Obig = (float*)alloc(hElems * sizeof(float));

  float* dis_mol = (float*)alloc((size_t)N_MOL * 4);
  float* dis_pro = (float*)alloc((size_t)N_PRO * 4);
  float* dis_ppi = (float*)alloc((size_t)B_PRO * 4);
  float* pool_mol= (float*)alloc((size_t)B_MOL * 128 * 4);
  float* cnt_mol = (float*)alloc((size_t)B_MOL * 4);
  float* pool_pro= (float*)alloc((size_t)B_PRO * 128 * 4);
  float* cnt_pro = (float*)alloc((size_t)B_PRO * 4);
  float* x_final = (float*)alloc((size_t)B_MOL * 128 * 4);
  float* p_final = (float*)alloc((size_t)B_PRO * 128 * 4);
  float* q_final = (float*)alloc((size_t)B_PRO * 128 * 4);
  float* fcH     = (float*)alloc((size_t)B_MOL * 1024 * 4);
  float* h2buf   = (float*)alloc((size_t)B_MOL * 512 * 4);
  bf16_t* actSm  = (bf16_t*)alloc((size_t)B_MOL * 1024 * sizeof(bf16_t));
  bf16_t* xcBF   = (bf16_t*)alloc((size_t)B_MOL * 256 * sizeof(bf16_t));

  auto g1 = [](long long n) { return dim3((unsigned)((n + 255) / 256)); };

  // weight -> bf16 WMMA B-fragment pack; N-tiles padded to a multiple of 8
  auto packW = [&](const float* W, int K, int N, int Kpad) -> bf16_t* {
    int Kt = Kpad / 32;
    int Nt = (((N + 15) / 16) + 7) & ~7;
    size_t elems = (size_t)Kt * Nt * 512;
    bf16_t* Wb = (bf16_t*)alloc(elems * sizeof(bf16_t));
    k_pack_wfrag<<<g1((long long)elems), 256, 0, stream>>>(W, Wb, K, N, Kt, Nt);
    return Wb;
  };

  bf16_t* Wb_mg1   = packW(w_mg1,   78, 156,   96);
  bf16_t* Wb_mg2   = packW(w_mg2,  156, 312,  160);
  bf16_t* Wb_mg3   = packW(w_mg3,  312, 128,  320);
  bf16_t* Wb_mfc1  = packW(w_mfc1, 128, 1024, 128);
  bf16_t* Wb_mfc2  = packW(w_mfc2, 1024, 128, 1024);
  bf16_t* Wb_pg1   = packW(w_pg1,   33, 128,   64);
  bf16_t* Wb_pg2   = packW(w_pg2,  128, 128,  128);
  bf16_t* Wb_pg3   = packW(w_pg3,  128, 128,  128);
  bf16_t* Wb_pfc1  = packW(w_pfc1, 128, 1024, 128);
  bf16_t* Wb_pfc2  = packW(w_pfc2, 1024, 128, 1024);
  bf16_t* Wb_ppig1 = packW(w_ppig1, 128, 1024, 128);
  bf16_t* Wb_ppig2 = packW(w_ppig2, 1024, 128, 1024);
  bf16_t* Wb_ppifc1= packW(w_ppifc1,128, 1024, 128);
  bf16_t* Wb_ppifc2= packW(w_ppifc2,1024, 128, 1024);
  bf16_t* Wb_fc1   = packW(w_fc1,  256, 1024, 256);
  bf16_t* Wb_fc2   = packW(w_fc2, 1024,  512, 1024);

  auto gemm = [&](const bf16_t* A, int M, int Kpad, const bf16_t* Wfrag,
                  const float* bias, float* C, int Nout, int relu) {
    int Nt = (((Nout + 15) / 16) + 7) & ~7;
    dim3 grid((unsigned)(Nt / 8), (unsigned)((M / 16 + 3) / 4));
    k_wmma_gemm<<<grid, 128, 0, stream>>>(A, M, Kpad, Wfrag, Kpad / 32,
                                          bias, C, Nout, relu);
  };

  auto gcn_layer = [&](const float* Xf32, int M, int K, int Kpad,
                       const bf16_t* Wfrag, const float* bias, int Nout,
                       const int* esrc, const int* edst, int E,
                       const float* dis, int reluIn,
                       bf16_t* actB, float* H, float* Out) {
    k_pack_act<<<g1((long long)M * Kpad), 256, 0, stream>>>(Xf32, actB, M, K, Kpad, reluIn);
    gemm(actB, M, Kpad, Wfrag, nullptr, H, Nout, 0);
    k_self_bias<<<g1((long long)M * Nout), 256, 0, stream>>>(H, dis, bias, Out, M, Nout);
    k_scatter4<<<g1((long long)E * (Nout >> 2)), 256, 0, stream>>>(esrc, edst, H, dis, Out, E, Nout);
  };

  auto fc = [&](const float* Xf32, int M, int K, const bf16_t* Wfrag,
                const float* bias, int Nout, int relu, int reluIn,
                bf16_t* actB, float* C) {
    k_pack_act<<<g1((long long)M * K), 256, 0, stream>>>(Xf32, actB, M, K, K, reluIn);
    gemm(actB, M, K, Wfrag, bias, C, Nout, relu);
  };

  // --- degree -> rsqrt (self loops give deg >= 1) ---
  k_fill<<<g1(N_MOL), 256, 0, stream>>>(dis_mol, 1.0f, N_MOL);
  k_deg <<<g1(E_MOL), 256, 0, stream>>>(mol_ei + E_MOL, dis_mol, E_MOL);
  k_rsqrt<<<g1(N_MOL), 256, 0, stream>>>(dis_mol, N_MOL);
  k_fill<<<g1(N_PRO), 256, 0, stream>>>(dis_pro, 1.0f, N_PRO);
  k_deg <<<g1(E_PRO), 256, 0, stream>>>(pro_ei + E_PRO, dis_pro, E_PRO);
  k_rsqrt<<<g1(N_PRO), 256, 0, stream>>>(dis_pro, N_PRO);
  k_fill<<<g1(B_PRO), 256, 0, stream>>>(dis_ppi, 1.0f, B_PRO);
  k_deg <<<g1(E_PPI), 256, 0, stream>>>(ppi_e + E_PPI, dis_ppi, E_PPI);
  k_rsqrt<<<g1(B_PRO), 256, 0, stream>>>(dis_ppi, B_PRO);

  const int* mol_s = mol_ei;           const int* mol_d = mol_ei + E_MOL;
  const int* pro_s = pro_ei;           const int* pro_d = pro_ei + E_PRO;
  const int* ppi_s = ppi_e;            const int* ppi_d = ppi_e + E_PPI;

  // ================= molecule branch =================
  gcn_layer(mol_x, N_MOL,  78,  96, Wb_mg1, b_mg1, 156, mol_s, mol_d, E_MOL, dis_mol, 0, actBig, Hbig, Obig);
  gcn_layer(Obig,  N_MOL, 156, 160, Wb_mg2, b_mg2, 312, mol_s, mol_d, E_MOL, dis_mol, 1, actBig, Hbig, Obig);
  gcn_layer(Obig,  N_MOL, 312, 320, Wb_mg3, b_mg3, 128, mol_s, mol_d, E_MOL, dis_mol, 1, actBig, Hbig, Obig);
  k_fill<<<g1((long long)B_MOL * 128), 256, 0, stream>>>(pool_mol, 0.0f, (long long)B_MOL * 128);
  k_fill<<<g1(B_MOL), 256, 0, stream>>>(cnt_mol, 0.0f, B_MOL);
  k_pool_accum<<<g1((long long)N_MOL * 128), 256, 0, stream>>>(Obig, mol_batch, pool_mol, cnt_mol, N_MOL, 128);
  k_pool_div<<<g1((long long)B_MOL * 128), 256, 0, stream>>>(pool_mol, cnt_mol, B_MOL, 128);
  fc(pool_mol, B_MOL, 128,  Wb_mfc1, b_mfc1, 1024, 1, 0, actSm, fcH);
  fc(fcH,      B_MOL, 1024, Wb_mfc2, b_mfc2,  128, 0, 0, actSm, x_final);

  // ================= protein branch =================
  gcn_layer(pro_x, N_PRO,  33,  64, Wb_pg1, b_pg1, 128, pro_s, pro_d, E_PRO, dis_pro, 0, actBig, Hbig, Obig);
  gcn_layer(Obig,  N_PRO, 128, 128, Wb_pg2, b_pg2, 128, pro_s, pro_d, E_PRO, dis_pro, 1, actBig, Hbig, Obig);
  gcn_layer(Obig,  N_PRO, 128, 128, Wb_pg3, b_pg3, 128, pro_s, pro_d, E_PRO, dis_pro, 1, actBig, Hbig, Obig);
  k_fill<<<g1((long long)B_PRO * 128), 256, 0, stream>>>(pool_pro, 0.0f, (long long)B_PRO * 128);
  k_fill<<<g1(B_PRO), 256, 0, stream>>>(cnt_pro, 0.0f, B_PRO);
  k_pool_accum<<<g1((long long)N_PRO * 128), 256, 0, stream>>>(Obig, pro_batch, pool_pro, cnt_pro, N_PRO, 128);
  k_pool_div<<<g1((long long)B_PRO * 128), 256, 0, stream>>>(pool_pro, cnt_pro, B_PRO, 128);
  fc(pool_pro, B_PRO, 128,  Wb_pfc1, b_pfc1, 1024, 1, 0, actSm, fcH);
  fc(fcH,      B_PRO, 1024, Wb_pfc2, b_pfc2,  128, 0, 0, actSm, p_final);

  // ================= PPI branch (graph over B_PRO nodes) =================
  gcn_layer(p_final, B_PRO, 128,  128,  Wb_ppig1, b_ppig1, 1024, ppi_s, ppi_d, E_PPI, dis_ppi, 0, actSm, Hbig, Obig);
  gcn_layer(Obig,    B_PRO, 1024, 1024, Wb_ppig2, b_ppig2,  128, ppi_s, ppi_d, E_PPI, dis_ppi, 1, actSm, Hbig, Obig);
  fc(Obig, B_PRO, 128,  Wb_ppifc1, b_ppifc1, 1024, 1, 1, actSm, fcH);   // reluIn: q = relu(gcn2)
  fc(fcH,  B_PRO, 1024, Wb_ppifc2, b_ppifc2,  128, 0, 0, actSm, q_final);

  // ================= head =================
  k_build_xc<<<g1((long long)B_MOL * 256), 256, 0, stream>>>(x_final, q_final, seq_num, xcBF, B_MOL);
  gemm(xcBF, B_MOL, 256, Wb_fc1, b_fc1, fcH, 1024, 1);
  k_pack_act<<<g1((long long)B_MOL * 1024), 256, 0, stream>>>(fcH, actSm, B_MOL, 1024, 1024, 0);
  gemm(actSm, B_MOL, 1024, Wb_fc2, b_fc2, h2buf, 512, 1);
  k_head_out<<<g1(B_MOL), 256, 0, stream>>>(h2buf, w_outp, b_outp, (float*)d_out, B_MOL, 512);
}
